// Attention_23682449670627
// MI455X (gfx1250) — matrix operand order repaired and checked by reading, not compiled, not run on hardware
//
#include <hip/hip_runtime.h>

typedef __attribute__((ext_vector_type(16))) __bf16 v16bf;
typedef __attribute__((ext_vector_type(8)))  float  v8f;
typedef __attribute__((ext_vector_type(4)))  int    v4i;
typedef __attribute__((ext_vector_type(8)))  int    v8i;

namespace {
constexpr int Bz   = 4;
constexpr int Sq   = 2048;
constexpr int Hd   = 576;
constexpr int NH   = 9;
constexpr int NKV  = 3;
constexpr int HD   = 64;
constexpr int NQKV = (NH + 2 * NKV) * HD;   // 960
constexpr int Mrows = Bz * Sq;              // 8192
constexpr float NEGV = -1.0e9f;
// 1/sqrt(HD) * log2(e), folded into Q so softmax runs in exp2 domain
constexpr float QSCALE = 0.18033688011112042f;
}

__device__ __forceinline__ float fast_exp2(float x) {
#if __has_builtin(__builtin_amdgcn_exp2f)
  return __builtin_amdgcn_exp2f(x);    // v_exp_f32
#else
  return exp2f(x);
#endif
}

// ---------------- WMMA operand loaders (wave32, contiguous k = 16*half + s per lane) ---

// 16-bit A matrix 16x32 (M x K), row-major source with stride lda (elements).
// lanes 0-15: M=lane, K=0..15; lanes 16-31: M=lane-16, K=16..31.
__device__ __forceinline__ v16bf load_a16x32(const __bf16* base, int lda) {
  const int lane = threadIdx.x & 31;
  const int m    = lane & 15;
  const int half = lane >> 4;
  const __bf16* row = base + (size_t)m * lda + half * 16;
  const v4i lo = *(const v4i*)row;          // K = half*16 + 0..7
  const v4i hi = *(const v4i*)(row + 8);    // K = half*16 + 8..15
  v8i raw;
#pragma unroll
  for (int i = 0; i < 4; ++i) { raw[i] = lo[i]; raw[i + 4] = hi[i]; }
  return __builtin_bit_cast(v16bf, raw);
}

// Pre-swizzled operand tile: 512 bf16 (1KB) stored lane-major, 32B per lane.
// Wave reads one contiguous 1KB block (2x b128 per lane). Used for packed
// B operands (weights, P) and packed A operands (V^T).
__device__ __forceinline__ v16bf load_packed(const __bf16* tile) {
  const v8i raw = *(const v8i*)(tile + (size_t)(threadIdx.x & 31) * 16);
  return __builtin_bit_cast(v16bf, raw);
}

// B matrix 32x16 where logical B[kk][n] = base[n*ld + kk] (transposed source,
// contiguous along the contraction dim). Used for Q^T.
__device__ __forceinline__ v16bf load_b32x16_T(const __bf16* base, int ld) {
  const int lane = threadIdx.x & 31;
  const int n    = lane & 15;
  const int half = lane >> 4;
  const v8i raw = *(const v8i*)(base + (size_t)n * ld + half * 16);
  return __builtin_bit_cast(v16bf, raw);
}

__device__ __forceinline__ v8f wmma_bf16(v16bf a, v16bf b, v8f c) {
  return __builtin_amdgcn_wmma_f32_16x16x32_bf16(false, a, false, b,
                                                 (short)0, c, false, false);
}

__device__ __forceinline__ v8f v8f_zero() {
  v8f z;
#pragma unroll
  for (int i = 0; i < 8; ++i) z[i] = 0.0f;
  return z;
}

// offset of element (k,n) inside a packed 32x16 B-operand tile
__device__ __forceinline__ int bswz(int k, int n) {
  return ((k >> 4) * 16 + n) * 16 + (k & 15);
}
// offset of element (m,kk) inside a packed 16x32 A-operand tile
__device__ __forceinline__ int aswz(int m, int kk) {
  return (m + 16 * (kk >> 4)) * 16 + (kk & 15);
}

// ---------------- conversion / packing --------------------------------------

__global__ void cvt_f32_bf16(const float* __restrict__ src,
                             __bf16* __restrict__ dst, int n) {
  int i = blockIdx.x * blockDim.x + threadIdx.x;
  if (i < n) dst[i] = (__bf16)src[i];
}

// pack [Wq|Wk|Wv] (Hd x NQKV) into B-operand tile order (1KB tiles)
__global__ void pack_wqkv_swz(const float* __restrict__ wq,
                              const float* __restrict__ wk,
                              const float* __restrict__ wv,
                              __bf16* __restrict__ w) {
  int i = blockIdx.x * blockDim.x + threadIdx.x;
  if (i >= Hd * NQKV) return;
  const int k = i / NQKV, n = i % NQKV;
  float v;
  if (n < NH * HD)              v = wq[(size_t)k * (NH * HD) + n];
  else if (n < (NH + NKV) * HD) v = wk[(size_t)k * (NKV * HD) + n - NH * HD];
  else                          v = wv[(size_t)k * (NKV * HD) + n - (NH + NKV) * HD];
  const size_t tile = (size_t)(k >> 5) * (NQKV / 16) + (n >> 4);
  w[tile * 512 + bswz(k & 31, n & 15)] = (__bf16)v;
}

// pack Wo (576 x 576) into B-operand tile order
__global__ void pack_wo_swz(const float* __restrict__ wo,
                            __bf16* __restrict__ w) {
  int i = blockIdx.x * blockDim.x + threadIdx.x;
  if (i >= (NH * HD) * Hd) return;
  const int k = i / Hd, n = i % Hd;
  const size_t tile = (size_t)(k >> 5) * (Hd / 16) + (n >> 4);
  w[tile * 512 + bswz(k & 31, n & 15)] = (__bf16)wo[i];
}

// ---------------- QKV projection + fused RoPE -------------------------------
// grid (Mrows/64, NQKV/64), block 128 (4 waves). Each wave: one 16x64 C tile.
__global__ void __launch_bounds__(128)
qkv_rope(const __bf16* __restrict__ xp, const __bf16* __restrict__ w,
         const float* __restrict__ cosb, const float* __restrict__ sinb,
         __bf16* __restrict__ qb, __bf16* __restrict__ kbuf,
         __bf16* __restrict__ vbuf) {
  const int lane = threadIdx.x & 31;
  const int wave = threadIdx.x >> 5;
  const int m0 = blockIdx.x * 64 + wave * 16;
  const int n0 = blockIdx.y * 64;            // 64-aligned -> spans one head

  v8f acc[4];
#pragma unroll
  for (int t = 0; t < 4; ++t) acc[t] = v8f_zero();

  for (int kk = 0; kk < Hd; kk += 32) {
    if (kk + 32 < Hd) {   // prefetch next K panel (global_prefetch_b8)
      __builtin_prefetch(xp + (size_t)(m0 + (lane & 15)) * Hd + kk + 32, 0, 3);
      __builtin_prefetch(w + ((size_t)((kk + 32) >> 5) * (NQKV / 16) + (n0 >> 4)) * 512
                           + (size_t)lane * 16, 0, 3);
    }
    const v16bf a = load_a16x32(xp + (size_t)m0 * Hd + kk, Hd);
    const __bf16* btile = w + ((size_t)(kk >> 5) * (NQKV / 16) + (n0 >> 4)) * 512;
#pragma unroll
    for (int t = 0; t < 4; ++t)
      acc[t] = wmma_bf16(a, load_packed(btile + t * 512), acc[t]);
  }

  const int half = lane >> 4;
  const int n    = lane & 15;
#pragma unroll
  for (int r = 0; r < 8; ++r) {
    const int m  = r + 8 * half;
    const int g  = m0 + m;
    const int bi = g / Sq;
    const int sp = g % Sq;
    const float* cr = cosb + (size_t)sp * HD;
    const float* sr = sinb + (size_t)sp * HD;
#pragma unroll
    for (int t = 0; t < 4; ++t) {
      const int d = t * 16 + n;              // head-dim index (n0 % 64 == 0)
      float outv = acc[t][r];
      if (n0 < (NH + NKV) * HD) {            // q or k head -> RoPE
        const int tp = (t < 2) ? t + 2 : t - 2;   // rotate-half partner d±32
        const float other = acc[tp][r];
        const float part  = (d < 32) ? -other : other;
        outv = outv * cr[d] + part * sr[d];
      }
      if (n0 < NH * HD) {                          // Q: scaled, row-major
        const int h = n0 / HD;
        qb[(((size_t)bi * NH + h) * Sq + sp) * HD + d] = (__bf16)(outv * QSCALE);
      } else if (n0 < (NH + NKV) * HD) {           // K: row-major
        const int h = (n0 - NH * HD) / HD;
        kbuf[(((size_t)bi * NKV + h) * Sq + sp) * HD + d] = (__bf16)outv;
      } else {                                     // V: A^T-operand tile order
        const int h = (n0 - (NH + NKV) * HD) / HD;
        const size_t tile = ((size_t)(bi * NKV + h) * (Sq / 32) + (sp >> 5)) * 4 + t;
        vbuf[tile * 512 + aswz(n, sp & 31)] = (__bf16)outv;
      }
    }
  }
}

// ---------------- causal flash attention (GQA), transposed scores -----------
// grid (Sq/32, NH, Bz), block 32 (one wave), 32 queries per wave.
// S^T = K*Q^T: each lane owns one query column -> softmax stats are in-register
// reductions + one shfl_xor(16). P lands in (k x q) layout = packed B-operand
// order for O^T = V^T * P. K and V operands are loaded ONCE per 32-key step
// and reused across both 16-query groups (halves load bytes per MAC).
__global__ void __launch_bounds__(32)
flash_attn(const __bf16* __restrict__ qb, const __bf16* __restrict__ kb,
           const __bf16* __restrict__ vb, __bf16* __restrict__ attn) {
  const int lane = threadIdx.x & 31;
  const int qt   = blockIdx.x;
  const int h    = blockIdx.y;
  const int bi   = blockIdx.z;
  const int hkv  = h / (NH / NKV);

  const int q0 = qt * 32;
  const __bf16* qptr  = qb + (((size_t)bi * NH + h) * Sq + q0) * HD;
  const __bf16* kbase = kb + (((size_t)bi * NKV + hkv) * Sq) * HD;
  const __bf16* vbase = vb + (size_t)(bi * NKV + hkv) * (Sq / 32) * 4 * 512;

  // Q^T B-operands: two query groups x two d-chunks, loaded once
  v16bf qT[2][2];
#pragma unroll
  for (int g = 0; g < 2; ++g) {
    qT[g][0] = load_b32x16_T(qptr + (size_t)g * 16 * HD, HD);
    qT[g][1] = load_b32x16_T(qptr + (size_t)g * 16 * HD + 32, HD);
  }

  v8f ot[2][4];                    // O^T accumulators, d = t*16 + r + 8*half
#pragma unroll
  for (int g = 0; g < 2; ++g)
#pragma unroll
    for (int t = 0; t < 4; ++t) ot[g][t] = v8f_zero();
  float mq[2] = {-3.0e38f, -3.0e38f};
  float lq[2] = {0.0f, 0.0f};

  __shared__ __bf16 pl[1024];      // P for both query groups (2x 1KB tiles)

  const int half  = lane >> 4;
  const int n     = lane & 15;
  const int nfull = qt;            // unmasked 32-key steps; then 1 masked step

  auto step = [&](int k0, bool masked) {
    if (!masked) {   // prefetch next K rows and V tiles
      __builtin_prefetch(kbase + (size_t)(k0 + 32 + n) * HD + half * 16, 0, 3);
      __builtin_prefetch(vbase + ((size_t)(k0 >> 5) + 1) * 4 * 512
                               + (size_t)lane * 16, 0, 3);
    }
    // K A-operands: loaded once, reused for both query groups
    v16bf ka[2][2];
#pragma unroll
    for (int j = 0; j < 2; ++j) {
      ka[j][0] = load_a16x32(kbase + (size_t)(k0 + j * 16) * HD, HD);
      ka[j][1] = load_a16x32(kbase + (size_t)(k0 + j * 16) * HD + 32, HD);
    }
#pragma unroll
    for (int g = 0; g < 2; ++g) {
      v8f st[2];
      st[0] = v8f_zero();
      st[1] = v8f_zero();
#pragma unroll
      for (int j = 0; j < 2; ++j) {   // S^T tile: 16 k-rows x 16 q-cols
        st[j] = wmma_bf16(ka[j][0], qT[g][0], st[j]);
        st[j] = wmma_bf16(ka[j][1], qT[g][1], st[j]);
      }
      if (masked) {
        const int qcol = q0 + g * 16 + n;
#pragma unroll
        for (int j = 0; j < 2; ++j)
#pragma unroll
          for (int r = 0; r < 8; ++r)
            if (k0 + j * 16 + r + 8 * half > qcol) st[j][r] = NEGV;
      }
      // per-lane softmax stats (lane owns query column within its half)
      float smax = st[0][0];
#pragma unroll
      for (int r = 1; r < 8; ++r) smax = fmaxf(smax, st[0][r]);
#pragma unroll
      for (int r = 0; r < 8; ++r) smax = fmaxf(smax, st[1][r]);
      smax = fmaxf(smax, __shfl_xor(smax, 16, 32));
      const float mnew  = fmaxf(mq[g], smax);
      const float alpha = fast_exp2(mq[g] - mnew);
      float rs = 0.0f;
#pragma unroll
      for (int j = 0; j < 2; ++j)
#pragma unroll
        for (int r = 0; r < 8; ++r) {
          const float p = fast_exp2(st[j][r] - mnew);
          st[j][r] = p;
          rs += p;
        }
      rs += __shfl_xor(rs, 16, 32);
      lq[g] = lq[g] * alpha + rs;
      mq[g] = mnew;
#pragma unroll
      for (int t = 0; t < 4; ++t) ot[g][t] = ot[g][t] * alpha;
      // P (k x q) -> LDS directly in packed B-operand order
#pragma unroll
      for (int j = 0; j < 2; ++j)
#pragma unroll
        for (int r = 0; r < 8; ++r)
          pl[g * 512 + (j * 16 + n) * 16 + r + 8 * half] = (__bf16)st[j][r];
    }
    __syncthreads();
    const v16bf pb0 = load_packed(pl);
    const v16bf pb1 = load_packed(pl + 512);
    const __bf16* vt = vbase + (size_t)(k0 >> 5) * 4 * 512;
#pragma unroll
    for (int t = 0; t < 4; ++t) {      // V^T loaded once, used by both groups
      const v16bf va = load_packed(vt + t * 512);
      ot[0][t] = wmma_bf16(va, pb0, ot[0][t]);
      ot[1][t] = wmma_bf16(va, pb1, ot[1][t]);
    }
    __syncthreads();
  };

  for (int kt = 0; kt < nfull; ++kt) step(kt * 32, false);
  step(nfull * 32, true);

#pragma unroll
  for (int g = 0; g < 2; ++g) {
    const float inv = 1.0f / lq[g];
    const size_t rowbase =
        ((size_t)bi * Sq + q0 + g * 16 + n) * (NH * HD) + (size_t)h * HD;
#pragma unroll
    for (int t = 0; t < 4; ++t)
#pragma unroll
      for (int r = 0; r < 8; ++r)
        attn[rowbase + t * 16 + r + 8 * half] = (__bf16)(ot[g][t][r] * inv);
  }
}

// ---------------- output projection -----------------------------------------
// grid (Mrows/64, Hd/64), block 128. 16x64-per-wave WMMA GEMM, f32 out.
__global__ void __launch_bounds__(128)
out_gemm(const __bf16* __restrict__ attn, const __bf16* __restrict__ wo,
         float* __restrict__ out) {
  const int lane = threadIdx.x & 31;
  const int wave = threadIdx.x >> 5;
  const int m0 = blockIdx.x * 64 + wave * 16;
  const int n0 = blockIdx.y * 64;

  v8f acc[4];
#pragma unroll
  for (int t = 0; t < 4; ++t) acc[t] = v8f_zero();

  for (int kk = 0; kk < NH * HD; kk += 32) {
    if (kk + 32 < NH * HD) {
      __builtin_prefetch(attn + (size_t)(m0 + (lane & 15)) * (NH * HD) + kk + 32, 0, 3);
      __builtin_prefetch(wo + ((size_t)((kk + 32) >> 5) * (Hd / 16) + (n0 >> 4)) * 512
                            + (size_t)lane * 16, 0, 3);
    }
    const v16bf a = load_a16x32(attn + (size_t)m0 * (NH * HD) + kk, NH * HD);
    const __bf16* btile = wo + ((size_t)(kk >> 5) * (Hd / 16) + (n0 >> 4)) * 512;
#pragma unroll
    for (int t = 0; t < 4; ++t)
      acc[t] = wmma_bf16(a, load_packed(btile + t * 512), acc[t]);
  }

  const int half = lane >> 4;
  const int n    = lane & 15;
#pragma unroll
  for (int r = 0; r < 8; ++r) {
    const int m = r + 8 * half;
#pragma unroll
    for (int t = 0; t < 4; ++t)
      out[(size_t)(m0 + m) * Hd + n0 + t * 16 + n] = acc[t][r];
  }
}

// ---------------- launch -----------------------------------------------------

extern "C" void kernel_launch(void* const* d_in, const int* in_sizes, int n_in,
                              void* d_out, int out_size, void* d_ws,
                              size_t ws_size, hipStream_t stream) {
  (void)in_sizes; (void)n_in; (void)out_size; (void)ws_size;
  const float* x    = (const float*)d_in[0];
  const float* cosb = (const float*)d_in[1];
  const float* sinb = (const float*)d_in[2];
  // d_in[3] is the dense (S,S) mask; causal mask is computed analytically.
  const float* wq = (const float*)d_in[4];
  const float* wk = (const float*)d_in[5];
  const float* wv = (const float*)d_in[6];
  const float* wo = (const float*)d_in[7];
  float* out = (float*)d_out;

  // workspace carve-up (~36.4 MB total)
  char* ws = (char*)d_ws;
  size_t off = 0;
  auto carve = [&](size_t bytes) {
    void* p = ws + off;
    off = (off + bytes + 255) & ~(size_t)255;
    return p;
  };
  __bf16* xp   = (__bf16*)carve((size_t)Mrows * Hd * 2);
  __bf16* wqkv = (__bf16*)carve((size_t)Hd * NQKV * 2);
  __bf16* wop  = (__bf16*)carve((size_t)(NH * HD) * Hd * 2);
  __bf16* qb   = (__bf16*)carve((size_t)Bz * NH * Sq * HD * 2);
  __bf16* kbuf = (__bf16*)carve((size_t)Bz * NKV * Sq * HD * 2);
  __bf16* vbuf = (__bf16*)carve((size_t)Bz * NKV * Sq * HD * 2);
  __bf16* attn = (__bf16*)carve((size_t)Mrows * (NH * HD) * 2);

  const int nx = Mrows * Hd;
  cvt_f32_bf16<<<(nx + 255) / 256, 256, 0, stream>>>(x, xp, nx);
  const int nwq = Hd * NQKV;
  pack_wqkv_swz<<<(nwq + 255) / 256, 256, 0, stream>>>(wq, wk, wv, wqkv);
  const int nwo = (NH * HD) * Hd;
  pack_wo_swz<<<(nwo + 255) / 256, 256, 0, stream>>>(wo, wop);

  qkv_rope<<<dim3(Mrows / 64, NQKV / 64), 128, 0, stream>>>(
      xp, wqkv, cosb, sinb, qb, kbuf, vbuf);
  flash_attn<<<dim3(Sq / 32, NH, Bz), 32, 0, stream>>>(qb, kbuf, vbuf, attn);
  out_gemm<<<dim3(Mrows / 64, Hd / 64), 128, 0, stream>>>(attn, wop, out);
}
